// BertClsLSTM_83416854823676
// MI455X (gfx1250) — matrix operand order repaired and checked by reading, not compiled, not run on hardware
//
#include <hip/hip_runtime.h>
#include <hip/hip_bf16.h>
#include <math.h>

// ---------------------------------------------------------------------------
// BertClsLSTM on MI455X (gfx1250): bf16 WMMA + TDM tensor DMA + async LDS loads.
//   B=512, L=128, H=768, LH=256, G=4*LH=1024, M=B*L=65536
// ---------------------------------------------------------------------------

typedef __bf16 bf16_t;
typedef __attribute__((ext_vector_type(16))) __bf16 v16bf;
typedef __attribute__((ext_vector_type(8)))  __bf16 v8bf;
typedef __attribute__((ext_vector_type(8)))  float  v8f;
typedef __attribute__((ext_vector_type(4)))  unsigned int u32x4;
typedef __attribute__((ext_vector_type(8)))  unsigned int u32x8;

#define LDS_PAD    8
#define LDS_STR    (32 + LDS_PAD)   // 80B row stride; matches TDM pad (16dw data + 4dw pad)

union FragU { v16bf v; v8bf h[2]; };

__device__ __forceinline__ v16bf load_frag(const bf16_t* lo, const bf16_t* hi) {
  FragU u;
  u.h[0] = *(const v8bf*)lo;
  u.h[1] = *(const v8bf*)hi;
  return u.v;
}

__device__ __forceinline__ unsigned lds_off(const void* p) {
  return (unsigned)(uintptr_t)p;   // generic LDS pointer: low 32 bits = LDS offset
}

// --- TDM: DMA one 128(rows) x 32(bf16) tile, row stride `stride_elems`, into LDS
//     with 16B pad per row (-> 80B LDS row stride == LDS_STR).  D# per ISA ch.8.
__device__ __forceinline__ void tdm_load_tile_2d(unsigned lds, const void* gaddr,
                                                 unsigned stride_elems) {
  unsigned long long ga = (unsigned long long)(uintptr_t)gaddr;
  u32x4 g0;
  g0[0] = 1u;                                              // count=1, user mode
  g0[1] = lds;                                             // lds_addr (bytes)
  g0[2] = (unsigned)ga;                                    // global_addr[31:0]
  g0[3] = ((unsigned)(ga >> 32) & 0x01FFFFFFu) | (2u << 30); // ga[56:32] | type=2
  u32x8 g1;
  g1[0] = (1u << 16)        // data_size = 2 bytes
        | (1u << 20)        // pad_enable
        | (3u << 22)        // pad_interval = 16 dwords (one 32-elem bf16 row)
        | (3u << 25);       // pad_amount   = 4 dwords (16B)
  g1[1] = (32u << 16);                    // tensor_dim0 lo16 (=32)
  g1[2] = (128u << 16);                   // tensor_dim0 hi | tensor_dim1 lo (=128)
  g1[3] = (32u << 16);                    // tensor_dim1 hi | tile_dim0 (=32)
  g1[4] = 128u;                           // tile_dim1 (=128) | tile_dim2 (=0)
  g1[5] = stride_elems;                   // tensor_dim0_stride lo32 (elements)
  g1[6] = 0u;
  g1[7] = 0u;
  asm volatile("tensor_load_to_lds %0, %1" :: "s"(g0), "s"(g1) : "memory");
}

// --- async per-lane 16B global->LDS copy (ASYNCcnt) ---
__device__ __forceinline__ void async_b128(unsigned lds, const void* g) {
  asm volatile("global_load_async_to_lds_b128 %0, %1, off"
               :: "v"(lds), "v"((unsigned long long)(uintptr_t)g) : "memory");
}
__device__ __forceinline__ void wait_async4() { asm volatile("s_wait_asynccnt 0x4" ::: "memory"); }
__device__ __forceinline__ void wait_async0() { asm volatile("s_wait_asynccnt 0x0" ::: "memory"); }

// ---------------------------------------------------------------------------
// K1: f32 -> bf16 conversion of x, w_ih, w_hh
// ---------------------------------------------------------------------------
__global__ void cvt_bf16_kernel(const float* __restrict__ x,
                                const float* __restrict__ wih,
                                const float* __restrict__ whh,
                                bf16_t* __restrict__ xbf,
                                bf16_t* __restrict__ wihbf,
                                bf16_t* __restrict__ whhbf) {
  const size_t NX = 512ull * 128 * 768;
  const size_t NI = 1024ull * 768;
  const size_t NH = 1024ull * 256;
  size_t i = (size_t)blockIdx.x * 256 + threadIdx.x;
  if (i < NX)                xbf[i]             = (bf16_t)x[i];
  else if (i < NX + NI)      wihbf[i - NX]      = (bf16_t)wih[i - NX];
  else if (i < NX + NI + NH) whhbf[i - NX - NI] = (bf16_t)whh[i - NX - NI];
}

// Conv weight transpose: f32 [Cout][Cin][K] -> bf16 [K][CoPad][Cin], zero-pad co>=Cout
__global__ void cvt_convw_kernel(const float* __restrict__ w, bf16_t* __restrict__ wt,
                                 int Cout, int CoPad, int Cin, int K) {
  size_t i = (size_t)blockIdx.x * 256 + threadIdx.x;
  size_t total = (size_t)K * CoPad * Cin;
  if (i >= total) return;
  int ci = (int)(i % Cin);
  int t  = (int)(i / Cin);
  int co = t % CoPad;
  int dk = t / CoPad;
  float v = (co < Cout) ? w[((size_t)co * Cin + ci) * K + dk] : 0.f;
  wt[i] = (bf16_t)v;
}

// Zero LSTM state: c (f32), hsum (f32), hbf (bf16)  -- 512*256 each
__global__ void init_state_kernel(float* __restrict__ c, float* __restrict__ hsum,
                                  bf16_t* __restrict__ hbf) {
  int i = blockIdx.x * 256 + threadIdx.x;  // 131072
  c[i] = 0.f; hsum[i] = 0.f; hbf[i] = (bf16_t)0.f;
}

// ---------------------------------------------------------------------------
// K2: XW[(l*512+b)][n] = x[m,:768] @ w_ih[n,:768]^T + (b_ih[n]+b_hh[n])
//     128x128x32 WMMA tiles; TDM double-buffered tile DMA. Grid (512,8), 256 thr.
// ---------------------------------------------------------------------------
__global__ __launch_bounds__(256) void gemm_xw_kernel(
    const bf16_t* __restrict__ A,   // xbf [65536,768]
    const bf16_t* __restrict__ Bw,  // wihbf [1024,768]
    const float* __restrict__ bih, const float* __restrict__ bhh,
    bf16_t* __restrict__ XW) {      // [(l*512+b)][1024]
  __shared__ bf16_t As[2][128][LDS_STR];
  __shared__ bf16_t Bs[2][128][LDS_STR];
  const int tid  = threadIdx.x;
  const int lane = tid & 31, wave = tid >> 5;
  const int lr = lane & 15, hi = lane >> 4;
  const int wm = (wave >> 1) * 32, wn = (wave & 1) * 64;
  const int m0 = blockIdx.x * 128, n0 = blockIdx.y * 128;
  v8f acc[2][4] = {};

  if (wave == 0) {
    tdm_load_tile_2d(lds_off(&As[0][0][0]), &A [(size_t)m0 * 768], 768u);
    tdm_load_tile_2d(lds_off(&Bs[0][0][0]), &Bw[(size_t)n0 * 768], 768u);
  }
  for (int k0 = 0; k0 < 768; k0 += 32) {
    const int buf = (k0 >> 5) & 1;
    if (wave == 0) {
      if (k0 + 32 < 768) {
        tdm_load_tile_2d(lds_off(&As[buf ^ 1][0][0]), &A [(size_t)m0 * 768 + k0 + 32], 768u);
        tdm_load_tile_2d(lds_off(&Bs[buf ^ 1][0][0]), &Bw[(size_t)n0 * 768 + k0 + 32], 768u);
        __builtin_amdgcn_s_wait_tensorcnt(2);   // current tile's 2 DMAs done
      } else {
        __builtin_amdgcn_s_wait_tensorcnt(0);
      }
    }
    __syncthreads();
    v16bf af[2], bfv[4];
    for (int mt = 0; mt < 2; ++mt) {
      int r = wm + mt * 16 + lr;
      af[mt] = load_frag(&As[buf][r][hi * 8], &As[buf][r][hi * 8 + 16]);
    }
    for (int nt = 0; nt < 4; ++nt) {
      int r = wn + nt * 16 + lr;
      bfv[nt] = load_frag(&Bs[buf][r][hi * 8], &Bs[buf][r][hi * 8 + 16]);
    }
    for (int mt = 0; mt < 2; ++mt)
      for (int nt = 0; nt < 4; ++nt)
        acc[mt][nt] = __builtin_amdgcn_wmma_f32_16x16x32_bf16(
            false, af[mt], false, bfv[nt], (short)0, acc[mt][nt], false, false);
    __syncthreads();
  }

  for (int mt = 0; mt < 2; ++mt) {
    int mbase = m0 + wm + mt * 16 + hi * 8;
    for (int nt = 0; nt < 4; ++nt) {
      int n = n0 + wn + nt * 16 + lr;
      float bias = bih[n] + bhh[n];
      for (int r = 0; r < 8; ++r) {
        int m = mbase + r;
        int b = m >> 7, l = m & 127;       // m = b*128 + l -> store time-major
        XW[((size_t)l * 512 + b) * 1024 + n] = (bf16_t)(acc[mt][nt][r] + bias);
      }
    }
  }
}

// ---------------------------------------------------------------------------
// K3: one LSTM step: gates[b][n] = XW[t][b][n] + hbf[b,:256] @ whh[n,:256]^T
//     Grid (4,8), 256 thr; TDM double-buffered tiles.
// ---------------------------------------------------------------------------
__global__ __launch_bounds__(256) void lstm_step_gemm_kernel(
    const bf16_t* __restrict__ Hbf,  // [512,256]
    const bf16_t* __restrict__ Whh,  // [1024,256]
    const bf16_t* __restrict__ XW,   // [(l*512+b)][1024]
    float* __restrict__ gates,       // [512,1024]
    int t) {
  __shared__ bf16_t As[2][128][LDS_STR];
  __shared__ bf16_t Bs[2][128][LDS_STR];
  const int tid  = threadIdx.x;
  const int lane = tid & 31, wave = tid >> 5;
  const int lr = lane & 15, hi = lane >> 4;
  const int wm = (wave >> 1) * 32, wn = (wave & 1) * 64;
  const int m0 = blockIdx.x * 128, n0 = blockIdx.y * 128;
  const bf16_t* C0 = XW + (size_t)t * 512 * 1024;

  if (wave == 0) {
    tdm_load_tile_2d(lds_off(&As[0][0][0]), &Hbf[(size_t)m0 * 256], 256u);
    tdm_load_tile_2d(lds_off(&Bs[0][0][0]), &Whh[(size_t)n0 * 256], 256u);
  }
  v8f acc[2][4];
  for (int mt = 0; mt < 2; ++mt)
    for (int nt = 0; nt < 4; ++nt)
      for (int r = 0; r < 8; ++r)
        acc[mt][nt][r] = (float)C0[(size_t)(m0 + wm + mt * 16 + hi * 8 + r) * 1024 +
                                   (n0 + wn + nt * 16 + lr)];

  for (int k0 = 0; k0 < 256; k0 += 32) {
    const int buf = (k0 >> 5) & 1;
    if (wave == 0) {
      if (k0 + 32 < 256) {
        tdm_load_tile_2d(lds_off(&As[buf ^ 1][0][0]), &Hbf[(size_t)m0 * 256 + k0 + 32], 256u);
        tdm_load_tile_2d(lds_off(&Bs[buf ^ 1][0][0]), &Whh[(size_t)n0 * 256 + k0 + 32], 256u);
        __builtin_amdgcn_s_wait_tensorcnt(2);
      } else {
        __builtin_amdgcn_s_wait_tensorcnt(0);
      }
    }
    __syncthreads();
    v16bf af[2], bfv[4];
    for (int mt = 0; mt < 2; ++mt) {
      int r = wm + mt * 16 + lr;
      af[mt] = load_frag(&As[buf][r][hi * 8], &As[buf][r][hi * 8 + 16]);
    }
    for (int nt = 0; nt < 4; ++nt) {
      int r = wn + nt * 16 + lr;
      bfv[nt] = load_frag(&Bs[buf][r][hi * 8], &Bs[buf][r][hi * 8 + 16]);
    }
    for (int mt = 0; mt < 2; ++mt)
      for (int nt = 0; nt < 4; ++nt)
        acc[mt][nt] = __builtin_amdgcn_wmma_f32_16x16x32_bf16(
            false, af[mt], false, bfv[nt], (short)0, acc[mt][nt], false, false);
    __syncthreads();
  }

  for (int mt = 0; mt < 2; ++mt)
    for (int nt = 0; nt < 4; ++nt) {
      int n = n0 + wn + nt * 16 + lr;
      for (int r = 0; r < 8; ++r) {
        int m = m0 + wm + mt * 16 + hi * 8 + r;
        gates[(size_t)m * 1024 + n] = acc[mt][nt][r];
      }
    }
}

// K4: elementwise LSTM cell (PyTorch gate order i,f,g,o), h-mean accumulation
__global__ void lstm_cell_kernel(const float* __restrict__ gates,
                                 float* __restrict__ cst,
                                 float* __restrict__ hsum,
                                 bf16_t* __restrict__ hbf) {
  int i = blockIdx.x * 256 + threadIdx.x;   // 131072 = 512*256
  int b = i >> 8, u = i & 255;
  const float* g = gates + (size_t)b * 1024 + u;
  float gi = g[0], gf = g[256], gg = g[512], go = g[768];
  float si = 1.f / (1.f + __expf(-gi));
  float sf = 1.f / (1.f + __expf(-gf));
  float so = 1.f / (1.f + __expf(-go));
  float c  = sf * cst[i] + si * tanhf(gg);
  float h  = so * tanhf(c);
  cst[i]  = c;
  hsum[i] += h;
  hbf[i]   = (bf16_t)h;
}

// ---------------------------------------------------------------------------
// K5: conv-as-GEMM with async im2col into LDS + bias + maxpool2(shfl) + relu.
//   IN  : bf16 [B][Lin][Cin], WT: bf16 [Kw][CoPad][Cin], OUT: bf16 [B][Lin/2][Cout]
//   Double-buffered global_load_async_to_lds_b128 (ASYNCcnt), 4 per thread/tile.
// ---------------------------------------------------------------------------
__global__ __launch_bounds__(256) void conv_wmma_kernel(
    const bf16_t* __restrict__ IN, const bf16_t* __restrict__ WT,
    const float* __restrict__ bias, bf16_t* __restrict__ OUT,
    int Lin, int Cin, int Kw, int Pad, int Cout, int CoPad) {
  __shared__ bf16_t As[2][128][LDS_STR];   // weights: rows = co
  __shared__ bf16_t Bs[2][128][LDS_STR];   // im2col:  rows = lp
  __shared__ __align__(16) bf16_t trash[8];
  const int tid  = threadIdx.x;
  const int lane = tid & 31, wave = tid >> 5;
  const int lr = lane & 15, hi = lane >> 4;
  const int wm = (wave >> 1) * 32, wn = (wave & 1) * 64;
  const int b   = blockIdx.x;
  const int co0 = blockIdx.z * 128;
  const int nk  = Cin >> 5;                // 32-wide ci chunks per tap
  const int KK  = Kw * nk;
  const int Lo  = Lin >> 1;
  v8f acc[2][4] = {};

  auto issue_tile = [&](int kk, int bufi) {
    int dk  = kk / nk;
    int ci0 = (kk - dk * nk) * 32;
    for (int i = 0; i < 2; ++i) {
      int c = tid + i * 256;               // 512 16B chunks per tile
      int row = c >> 2, off = (c & 3) * 8;
      async_b128(lds_off(&As[bufi][row][off]),
                 &WT[((size_t)dk * CoPad + co0 + row) * Cin + ci0 + off]);
      int lin = row + dk - Pad;
      bool ok = (row < Lin) && (lin >= 0) && (lin < Lin);
      unsigned      ldst = ok ? lds_off(&Bs[bufi][row][off]) : lds_off(&trash[0]);
      const bf16_t* gsrc = ok ? &IN[((size_t)b * Lin + lin) * Cin + ci0 + off] : IN;
      async_b128(ldst, gsrc);              // uniform ASYNCcnt: OOB lanes -> trash
      if (!ok) *(v8bf*)&Bs[bufi][row][off] = (v8bf){};   // zero-fill real slot
    }
  };

  issue_tile(0, 0);
  for (int kk = 0; kk < KK; ++kk) {
    const int buf = kk & 1;
    if (kk + 1 < KK) { issue_tile(kk + 1, buf ^ 1); wait_async4(); }
    else             { wait_async0(); }
    __syncthreads();
    v16bf af[2], bfv[4];
    for (int mt = 0; mt < 2; ++mt) {
      int r = wm + mt * 16 + lr;
      af[mt] = load_frag(&As[buf][r][hi * 8], &As[buf][r][hi * 8 + 16]);
    }
    for (int nt = 0; nt < 4; ++nt) {
      int r = wn + nt * 16 + lr;
      bfv[nt] = load_frag(&Bs[buf][r][hi * 8], &Bs[buf][r][hi * 8 + 16]);
    }
    for (int mt = 0; mt < 2; ++mt)
      for (int nt = 0; nt < 4; ++nt)
        acc[mt][nt] = __builtin_amdgcn_wmma_f32_16x16x32_bf16(
            false, af[mt], false, bfv[nt], (short)0, acc[mt][nt], false, false);
    __syncthreads();
  }

  for (int mt = 0; mt < 2; ++mt) {
    int cobase = co0 + wm + mt * 16 + hi * 8;
    for (int nt = 0; nt < 4; ++nt) {
      int n = wn + nt * 16 + lr;          // lp position (lane dim)
      for (int r = 0; r < 8; ++r) {
        int co = cobase + r;
        float v = acc[mt][nt][r] + ((co < Cout) ? bias[co] : 0.f);
        float mx = fmaxf(v, __shfl_xor(v, 1, 32));   // maxpool over (lp, lp^1)
        float rv = fmaxf(mx, 0.f);
        if (((lr & 1) == 0) && co < Cout && n < Lin) {
          int lo = n >> 1;
          OUT[((size_t)b * Lo + lo) * Cout + co] = (bf16_t)rv;
        }
      }
    }
  }
}

// K6: conv4 (1x1, 256->16) + relu   (tiny tail -> VALU)
__global__ void conv4_kernel(const bf16_t* __restrict__ Y3,  // [B][16][256]
                             const float* __restrict__ W4,   // [16][256]
                             const float* __restrict__ B4,
                             float* __restrict__ Y4) {       // [B][16(l)][16(co)]
  int i = blockIdx.x * 256 + threadIdx.x;  // 131072
  int co = i & 15, l = (i >> 4) & 15, b = i >> 8;
  const bf16_t* yp = Y3 + ((size_t)b * 16 + l) * 256;
  const float*  wp = W4 + co * 256;
  float s = B4[co];
  for (int k = 0; k < 256; ++k) s += (float)yp[k] * wp[k];
  Y4[i] = fmaxf(s, 0.f);
}

// K7: concat(hsum/128, cnn-features) -> fc1 -> fc2 -> fc3 (all relu)
__global__ __launch_bounds__(128) void mlp_head_kernel(
    const float* __restrict__ hsum, const float* __restrict__ Y4,
    const float* __restrict__ f1w, const float* __restrict__ f1b,
    const float* __restrict__ f2w, const float* __restrict__ f2b,
    const float* __restrict__ f3w, const float* __restrict__ f3b,
    float* __restrict__ out) {
  __shared__ float feat[512];
  __shared__ float z1[128];
  __shared__ float z2[32];
  int b = blockIdx.x, t = threadIdx.x;
  for (int i = t; i < 512; i += 128) {
    float v;
    if (i < 256) v = hsum[(size_t)b * 256 + i] * (1.f / 128.f);
    else {
      int f = i - 256, c = f >> 4, l = f & 15;       // ref flatten: c*16 + l
      v = Y4[((size_t)b * 16 + l) * 16 + c];
    }
    feat[i] = v;
  }
  __syncthreads();
  float s = f1b[t];
  for (int k = 0; k < 512; ++k) s += feat[k] * f1w[(size_t)t * 512 + k];
  z1[t] = fmaxf(s, 0.f);
  __syncthreads();
  if (t < 32) {
    float s2 = f2b[t];
    for (int k = 0; k < 128; ++k) s2 += z1[k] * f2w[(size_t)t * 128 + k];
    z2[t] = fmaxf(s2, 0.f);
  }
  __syncthreads();
  if (t < 2) {
    float s3 = f3b[t];
    for (int k = 0; k < 32; ++k) s3 += z2[k] * f3w[(size_t)t * 32 + k];
    out[b * 2 + t] = fmaxf(s3, 0.f);
  }
}

// ---------------------------------------------------------------------------
extern "C" void kernel_launch(void* const* d_in, const int* in_sizes, int n_in,
                              void* d_out, int out_size, void* d_ws, size_t ws_size,
                              hipStream_t stream) {
  (void)in_sizes; (void)n_in; (void)out_size; (void)ws_size;
  const float* x    = (const float*)d_in[0];
  const float* wih  = (const float*)d_in[1];
  const float* whh  = (const float*)d_in[2];
  const float* bih  = (const float*)d_in[3];
  const float* bhh  = (const float*)d_in[4];
  const float* c1w  = (const float*)d_in[5];
  const float* c1b  = (const float*)d_in[6];
  const float* c2w  = (const float*)d_in[7];
  const float* c2b  = (const float*)d_in[8];
  const float* c3w  = (const float*)d_in[9];
  const float* c3b  = (const float*)d_in[10];
  const float* c4w  = (const float*)d_in[11];
  const float* c4b  = (const float*)d_in[12];
  const float* f1w  = (const float*)d_in[13];
  const float* f1b  = (const float*)d_in[14];
  const float* f2w  = (const float*)d_in[15];
  const float* f2b  = (const float*)d_in[16];
  const float* f3w  = (const float*)d_in[17];
  const float* f3b  = (const float*)d_in[18];
  float* out = (float*)d_out;

  char* p = (char*)d_ws;
  auto carve = [&](size_t bytes) { char* r = p; p += (bytes + 255) & ~(size_t)255; return r; };
  const size_t NX = 512ull * 128 * 768;
  bf16_t* xbf   = (bf16_t*)carve(NX * 2);                     // 96 MB
  bf16_t* wihbf = (bf16_t*)carve(1024ull * 768 * 2);
  bf16_t* whhbf = (bf16_t*)carve(1024ull * 256 * 2);
  bf16_t* XW    = (bf16_t*)carve(65536ull * 1024 * 2);        // 128 MB, time-major
  float*  gates = (float*) carve(512ull * 1024 * 4);
  float*  cst   = (float*) carve(512ull * 256 * 4);
  float*  hsum  = (float*) carve(512ull * 256 * 4);
  bf16_t* hbf   = (bf16_t*)carve(512ull * 256 * 2);
  bf16_t* wt1   = (bf16_t*)carve(7ull * 256 * 768 * 2);
  bf16_t* wt2   = (bf16_t*)carve(5ull * 128 * 256 * 2);
  bf16_t* wt3   = (bf16_t*)carve(3ull * 256 * 64 * 2);
  bf16_t* y1    = (bf16_t*)carve(512ull * 64 * 256 * 2);
  bf16_t* y2    = (bf16_t*)carve(512ull * 32 * 64 * 2);
  bf16_t* y3    = (bf16_t*)carve(512ull * 16 * 256 * 2);
  float*  y4    = (float*) carve(512ull * 16 * 16 * 4);

  {
    size_t tot = NX + 1024ull * 768 + 1024ull * 256;
    cvt_bf16_kernel<<<dim3((unsigned)((tot + 255) / 256)), 256, 0, stream>>>(
        x, wih, whh, xbf, wihbf, whhbf);
  }
  cvt_convw_kernel<<<dim3((7u * 256 * 768 + 255) / 256), 256, 0, stream>>>(c1w, wt1, 256, 256, 768, 7);
  cvt_convw_kernel<<<dim3((5u * 128 * 256 + 255) / 256), 256, 0, stream>>>(c2w, wt2,  64, 128, 256, 5);
  cvt_convw_kernel<<<dim3((3u * 256 *  64 + 255) / 256), 256, 0, stream>>>(c3w, wt3, 256, 256,  64, 3);
  init_state_kernel<<<512, 256, 0, stream>>>(cst, hsum, hbf);

  gemm_xw_kernel<<<dim3(512, 8), 256, 0, stream>>>(xbf, wihbf, bih, bhh, XW);

  for (int t = 0; t < 128; ++t) {
    lstm_step_gemm_kernel<<<dim3(4, 8), 256, 0, stream>>>(hbf, whhbf, XW, gates, t);
    lstm_cell_kernel<<<512, 256, 0, stream>>>(gates, cst, hsum, hbf);
  }

  conv_wmma_kernel<<<dim3(512, 1, 2), 256, 0, stream>>>(xbf, wt1, c1b, y1, 128, 768, 7, 3, 256, 256);
  conv_wmma_kernel<<<dim3(512, 1, 1), 256, 0, stream>>>(y1,  wt2, c2b, y2,  64, 256, 5, 2,  64, 128);
  conv_wmma_kernel<<<dim3(512, 1, 2), 256, 0, stream>>>(y2,  wt3, c3b, y3,  32,  64, 3, 1, 256, 256);
  conv4_kernel<<<512, 256, 0, stream>>>(y3, c4w, c4b, y4);

  mlp_head_kernel<<<512, 128, 0, stream>>>(hsum, y4, f1w, f1b, f2w, f2b, f3w, f3b, out);
}
